// CAMPS_42872363548986
// MI455X (gfx1250) — matrix operand order repaired and checked by reading, not compiled, not run on hardware
//
#include <hip/hip_runtime.h>

#define N_NODES 8192
#define N_EDGES 131072
#define NB      256
#define F_INC   64
#define NHID    128
#define H3      384      // 3*NH
#define D6      768      // 6*NH
#define QSD     1536     // 2*D6
#define G4      3072     // 4*D6
#define KCAT    2304     // QSD + D6
#define CH      64       // per-graph row chunk for masked sim

typedef __bf16 bf16;
typedef __attribute__((ext_vector_type(16))) __bf16 v16bf;
typedef __attribute__((ext_vector_type(8)))  float  v8f;
typedef __attribute__((ext_vector_type(4)))  int    v4i;

#define AS_GLOBAL __attribute__((address_space(1)))
#define AS_LDS    __attribute__((address_space(3)))

#if __has_builtin(__builtin_amdgcn_global_load_async_to_lds_b128) && \
    __has_builtin(__builtin_amdgcn_s_wait_asynccnt)
#define HAVE_ASYNC_LDS 1
#else
#define HAVE_ASYNC_LDS 0
#endif

// ---------------------------------------------------------------- WMMA core
__device__ inline v8f wmma_bf16(v16bf a, v16bf b, v8f c) {
  // D = A(16x32) * B(32x16) + C, f32 accumulate
  return __builtin_amdgcn_wmma_f32_16x16x32_bf16(
      /*neg_a=*/false, a, /*neg_b=*/false, b,
      /*c_mod=*/(short)0, c, /*reuse_a=*/false, /*reuse_b=*/false);
}

// A fragment: row-major A[m][k], tile origin (m0,k0), 16x32 bf16.
// lane m = lane&15; VGPR v holds K = (lane>>4)*16 + v*2 (+1)
__device__ inline v16bf frag_a(const bf16* A, int lda, int m0, int k0) {
  int lane = threadIdx.x & 31;
  const bf16* row = A + (size_t)(m0 + (lane & 15)) * lda + k0;
  int h8 = (lane >> 4) * 16;
  v16bf r;
#pragma unroll
  for (int v = 0; v < 8; ++v) {
    int kb = h8 + v * 2;
    r[2 * v]     = row[kb];
    r[2 * v + 1] = row[kb + 1];
  }
  return r;
}

// A fragment where logical A = S^T (S row-major [k][m]); A(mm,kk) = S[kk][mm]
__device__ inline v16bf frag_a_t(const bf16* S, int lds_, int m0, int k0) {
  int lane = threadIdx.x & 31;
  int mm = m0 + (lane & 15);
  int h8 = (lane >> 4) * 16;
  v16bf r;
#pragma unroll
  for (int v = 0; v < 8; ++v) {
    int kb = h8 + v * 2;
    r[2 * v]     = S[(size_t)(k0 + kb) * lds_ + mm];
    r[2 * v + 1] = S[(size_t)(k0 + kb + 1) * lds_ + mm];
  }
  return r;
}

// B fragment for B = X^T where X is row-major [n][k]: B(kk,nn) = X[nn][kk]
// lanes 0-15: n=lane, K=0..15 ; lanes 16-31: n=lane-16, K=16..31 (2 packed per VGPR)
__device__ inline v16bf frag_b_t(const bf16* X, int ldx, int n0, int k0) {
  int lane = threadIdx.x & 31;
  const bf16* row = X + (size_t)(n0 + (lane & 15)) * ldx + k0 + ((lane >> 4) * 16);
  v16bf r;
#pragma unroll
  for (int v = 0; v < 8; ++v) { r[2 * v] = row[2 * v]; r[2 * v + 1] = row[2 * v + 1]; }
  return r;
}

// B fragment for row-major B[k][n]
__device__ inline v16bf frag_b_n(const bf16* B, int ldb, int k0, int n0) {
  int lane = threadIdx.x & 31;
  int n = n0 + (lane & 15);
  int kh = (lane >> 4) * 16;
  v16bf r;
#pragma unroll
  for (int v = 0; v < 8; ++v) {
    int k = k0 + kh + 2 * v;
    r[2 * v]     = B[(size_t)k * ldb + n];
    r[2 * v + 1] = B[(size_t)(k + 1) * ldb + n];
  }
  return r;
}

// store 16x16 f32 accumulator as bf16 into row-major S tile (m0,n0)
__device__ inline void store_d_bf(bf16* S, int lds_, int m0, int n0, v8f d) {
  int lane = threadIdx.x & 31;
  int n = n0 + (lane & 15);
  int mh = (lane >> 4) * 8;
#pragma unroll
  for (int v = 0; v < 8; ++v) S[(size_t)(m0 + mh + v) * lds_ + n] = (bf16)d[v];
}

// accumulate 16x16 f32 tile into global C (+=), guarding rows < mlim
__device__ inline void accum_d_global(float* C, int ldc, int m0, int n0, int mlim, v8f d) {
  int lane = threadIdx.x & 31;
  int n = n0 + (lane & 15);
  int mh = (lane >> 4) * 8;
#pragma unroll
  for (int v = 0; v < 8; ++v) {
    int m = m0 + mh + v;
    if (m < mlim) C[(size_t)m * ldc + n] += d[v];
  }
}

// Stage `rows` contiguous rows of src (row-major, H3 bf16/row) into LDS chunk,
// zero-padding rows [rows, CH). Uses CDNA5 async global->LDS DMA when available.
__device__ inline void stage_chunk(bf16* lds, const bf16* src, int rows, int tid, int nthr) {
#if HAVE_ASYNC_LDS
  // valid region is contiguous: rows*H3 bf16 = rows*48 x 16-byte beats
  int nvec = rows * (H3 * 2 / 16);
  for (int v = tid; v < nvec; v += nthr) {
    const char* g = (const char*)src + (size_t)v * 16;
    char* l = (char*)lds + (size_t)v * 16;
    __builtin_amdgcn_global_load_async_to_lds_b128(
        (AS_GLOBAL v4i*)(v4i*)g, (AS_LDS v4i*)(v4i*)l, 0, 0);
  }
  for (int idx = rows * H3 + tid; idx < CH * H3; idx += nthr) lds[idx] = (bf16)0.f;
#else
  for (int idx = tid; idx < CH * H3; idx += nthr) {
    int r = idx / H3, c = idx % H3;
    lds[idx] = (r < rows) ? src[(size_t)r * H3 + c] : (bf16)0.f;
  }
#endif
}

__device__ inline void stage_wait() {
#if HAVE_ASYNC_LDS
  __builtin_amdgcn_s_wait_asynccnt(0);
#endif
  __syncthreads();
}

// ------------------------------------------------------------- small kernels
__global__ void fill_f32(float* p, float v, int n) {
  int i = blockIdx.x * blockDim.x + threadIdx.x;
  if (i < n) p[i] = v;
}

__global__ void offsets_kernel(const int* __restrict__ batch, int n, int* __restrict__ off) {
  int b = blockIdx.x * blockDim.x + threadIdx.x;
  if (b > NB) return;
  int lo = 0, hi = n;                 // lower_bound(batch, b) ; batch is sorted
  while (lo < hi) { int mid = (lo + hi) >> 1; if (batch[mid] < b) lo = mid + 1; else hi = mid; }
  off[b] = lo;
}

__global__ void deg_acc(const int* __restrict__ ei, const float* __restrict__ ew,
                        float* __restrict__ deg, int E) {
  int e = blockIdx.x * blockDim.x + threadIdx.x;
  if (e < E) atomicAdd(&deg[ei[E + e]], ew[e]);
}

__global__ void rsqrt_inplace(float* p, int n) {
  int i = blockIdx.x * blockDim.x + threadIdx.x;
  if (i < n) p[i] = rsqrtf(p[i]);
}

// xw[n][128] = X[n][0:K] @ W[K][128]  (f32, one node row per block)
__global__ void gemm_xw(const float* __restrict__ X, int ldx,
                        const float* __restrict__ W, int K, float* __restrict__ out) {
  __shared__ float xr[NHID];
  int n = blockIdx.x, c = threadIdx.x;
  for (int k = c; k < K; k += NHID) xr[k] = X[(size_t)n * ldx + k];
  __syncthreads();
  float acc = 0.f;
  for (int k = 0; k < K; ++k) acc = fmaf(xr[k], W[k * NHID + c], acc);
  out[(size_t)n * NHID + c] = acc;
}

__global__ void gcn_self(const float* __restrict__ xw, const float* __restrict__ dinv,
                         float* __restrict__ hcol, int ldh) {
  int i = blockIdx.x, c = threadIdx.x;
  float di = dinv[i];
  hcol[(size_t)i * ldh + c] = xw[(size_t)i * NHID + c] * di * di;
}

__global__ void gcn_edge(const int* __restrict__ ei, const float* __restrict__ ew,
                         const float* __restrict__ xw, const float* __restrict__ dinv,
                         float* __restrict__ hcol, int ldh, int E) {
  int e = blockIdx.x, c = threadIdx.x;
  int s = ei[e], d = ei[E + e];
  float w = ew[e] * dinv[s] * dinv[d];
  atomicAdd(&hcol[(size_t)d * ldh + c], xw[(size_t)s * NHID + c] * w);
}

__global__ void gcn_finish(float* __restrict__ hcol, int ldh, const float* __restrict__ bias) {
  int i = blockIdx.x, c = threadIdx.x;
  float v = hcol[(size_t)i * ldh + c] + bias[c];
  hcol[(size_t)i * ldh + c] = (v > 0.f) ? v : 0.2f * v;   // leaky_relu 0.2
}

// L2-normalize h[i][0:384] -> f[i][0:384] (f32) + nbf[i][0:384] (bf16); zero f[i][384:768]
__global__ void l2norm_kernel(const float* __restrict__ h, float* __restrict__ f,
                              bf16* __restrict__ nbf) {
  __shared__ float red[128];
  int i = blockIdx.x, t = threadIdx.x;
  float s = 0.f;
  for (int c = t; c < H3; c += 128) { float v = h[(size_t)i * H3 + c]; s += v * v; }
  red[t] = s; __syncthreads();
  for (int o = 64; o > 0; o >>= 1) { if (t < o) red[t] += red[t + o]; __syncthreads(); }
  float inv = 1.f / fmaxf(sqrtf(red[0]), 1e-12f);
  for (int c = t; c < H3; c += 128) {
    float v = h[(size_t)i * H3 + c] * inv;
    f[(size_t)i * D6 + c] = v;
    f[(size_t)i * D6 + H3 + c] = 0.f;
    nbf[(size_t)i * H3 + c] = (bf16)v;
  }
}

// -------------------------------------------- masked interaction (block-diag)
// per graph b: S = ns_b @ nv_b^T ; f1[:,384:768] += S @ nv_b ; f2[:,384:768] += S^T @ ns_b
__global__ void masked_sim_kernel(const bf16* __restrict__ nsbf, const bf16* __restrict__ nvbf,
                                  const int* __restrict__ off1, const int* __restrict__ off2,
                                  float* __restrict__ f1, float* __restrict__ f2) {
  extern __shared__ char smem[];
  bf16* ldsA = (bf16*)smem;                 // [CH][H3]
  bf16* ldsB = ldsA + CH * H3;              // [CH][H3]
  bf16* ldsS = ldsB + CH * H3;              // [CH][CH]
  int b = blockIdx.x;
  int r1s = off1[b], m = off1[b + 1] - r1s;
  int r2s = off2[b], n = off2[b + 1] - r2s;
  if (m <= 0 || n <= 0) return;
  int tid = threadIdx.x, wave = tid >> 5, nthr = blockDim.x;

  for (int mc = 0; mc < m; mc += CH) {
    int mi = min(CH, m - mc);
    stage_chunk(ldsA, nsbf + (size_t)(r1s + mc) * H3, mi, tid, nthr);
    for (int nc = 0; nc < n; nc += CH) {
      int ni = min(CH, n - nc);
      stage_chunk(ldsB, nvbf + (size_t)(r2s + nc) * H3, ni, tid, nthr);
      stage_wait();
      // S(64x64) = A @ B^T ; 16 tiles over 8 waves
      for (int t = wave; t < 16; t += 8) {
        int tm = (t >> 2) * 16, tn = (t & 3) * 16;
        v8f acc = {};
#pragma unroll
        for (int k = 0; k < H3; k += 32)
          acc = wmma_bf16(frag_a(ldsA, H3, tm, k), frag_b_t(ldsB, H3, tn, k), acc);
        store_d_bf(ldsS, CH, tm, tn, acc);
      }
      __syncthreads();
      // p1 += S @ Bchunk : 4 x 24 tiles
      for (int t = wave; t < 4 * 24; t += 8) {
        int tm = (t / 24) * 16, tn = (t % 24) * 16;
        v8f acc = {};
#pragma unroll
        for (int k = 0; k < CH; k += 32)
          acc = wmma_bf16(frag_a(ldsS, CH, tm, k), frag_b_n(ldsB, H3, k, tn), acc);
        accum_d_global(f1 + (size_t)(r1s + mc) * D6 + H3, D6, tm, tn, mi, acc);
      }
      // p2 += S^T @ Achunk : 4 x 24 tiles
      for (int t = wave; t < 4 * 24; t += 8) {
        int tm = (t / 24) * 16, tn = (t % 24) * 16;
        v8f acc = {};
#pragma unroll
        for (int k = 0; k < CH; k += 32)
          acc = wmma_bf16(frag_a_t(ldsS, CH, tm, k), frag_b_n(ldsA, H3, k, tn), acc);
        accum_d_global(f2 + (size_t)(r2s + nc) * D6 + H3, D6, tm, tn, ni, acc);
      }
      __syncthreads();
    }
  }
}

// ------------------------------------------------- generic WMMA GEMM: C=A@Bt^T
// A: bf16 [M][K] row-major ; Bt: bf16 [N][K] row-major (torch weight layout)
// grid.x = M/16 ; grid.y = N/128 ; 256 threads = 8 waves, one 16x16 tile each
__global__ __launch_bounds__(256)
void wmma_gemm_abt(const bf16* __restrict__ A, int lda, const bf16* __restrict__ Bt, int ldb,
                   float* __restrict__ C, int ldc, int K) {
  int mt = blockIdx.x * 16;
  int nt = (blockIdx.y * 8 + (threadIdx.x >> 5)) * 16;
  v8f acc = {};
  for (int k = 0; k < K; k += 32) {
    __builtin_prefetch(&A[(size_t)mt * lda + k + 128], 0, 1);
    __builtin_prefetch(&Bt[(size_t)nt * ldb + k + 128], 0, 1);
    acc = wmma_bf16(frag_a(A, lda, mt, k), frag_b_t(Bt, ldb, nt, k), acc);
  }
  int lane = threadIdx.x & 31;
  int n = nt + (lane & 15);
  int mh = (lane >> 4) * 8;
#pragma unroll
  for (int v = 0; v < 8; ++v) C[(size_t)(mt + mh + v) * ldc + n] = acc[v];
}

// ----------------------------------------------------------- set2set kernels
__global__ void wcat_kernel(const float* __restrict__ Wih, const float* __restrict__ Whh,
                            bf16* __restrict__ Wc) {
  int idx = blockIdx.x * blockDim.x + threadIdx.x;
  if (idx >= G4 * KCAT) return;
  int o = idx / KCAT, k = idx % KCAT;
  float v = (k < QSD) ? Wih[(size_t)o * QSD + k] : Whh[(size_t)o * D6 + (k - QSD)];
  Wc[idx] = (bf16)v;
}

__global__ void f2bf_kernel(const float* __restrict__ s, bf16* __restrict__ d, int n) {
  int i = blockIdx.x * blockDim.x + threadIdx.x;
  if (i < n) d[i] = (bf16)s[i];
}

__global__ void qh_prep(const float* __restrict__ qs, const float* __restrict__ h,
                        bf16* __restrict__ qh) {
  int idx = blockIdx.x * blockDim.x + threadIdx.x;
  if (idx >= NB * KCAT) return;
  int b = idx / KCAT, k = idx % KCAT;
  qh[idx] = (bf16)((k < QSD) ? qs[(size_t)b * QSD + k] : h[(size_t)b * D6 + (k - QSD)]);
}

__device__ inline float sigm(float x) { return 1.f / (1.f + expf(-x)); }

__global__ void lstm_update(const float* __restrict__ g, const float* __restrict__ bih,
                            const float* __restrict__ bhh, float* __restrict__ h,
                            float* __restrict__ c) {
  int idx = blockIdx.x * blockDim.x + threadIdx.x;
  if (idx >= NB * D6) return;
  int b = idx / D6, j = idx % D6;
  const float* gb = g + (size_t)b * G4;
  float gi = gb[j]            + bih[j]            + bhh[j];
  float gf = gb[D6 + j]       + bih[D6 + j]       + bhh[D6 + j];
  float gg = gb[2 * D6 + j]   + bih[2 * D6 + j]   + bhh[2 * D6 + j];
  float go = gb[3 * D6 + j]   + bih[3 * D6 + j]   + bhh[3 * D6 + j];
  float cc = sigm(gf) * c[idx] + sigm(gi) * tanhf(gg);
  c[idx] = cc;
  h[idx] = sigm(go) * tanhf(cc);
}

__global__ void e_dot(const float* __restrict__ x, const float* __restrict__ h,
                      const int* __restrict__ batch, float* __restrict__ e) {
  __shared__ float red[256];
  int i = blockIdx.x, t = threadIdx.x;
  int b = batch[i];
  float s = 0.f;
  for (int c = t; c < D6; c += 256) s += x[(size_t)i * D6 + c] * h[(size_t)b * D6 + c];
  red[t] = s; __syncthreads();
  for (int o = 128; o > 0; o >>= 1) { if (t < o) red[t] += red[t + o]; __syncthreads(); }
  if (t == 0) e[i] = red[0];
}

__global__ void seg_softmax(float* __restrict__ e, const int* __restrict__ off) {
  __shared__ float red[256];
  __shared__ float stat[2];
  int b = blockIdx.x, t = threadIdx.x;
  int s0 = off[b], s1 = off[b + 1];
  if (s1 <= s0) return;
  float mx = -3.402823466e38f;
  for (int i = s0 + t; i < s1; i += 256) mx = fmaxf(mx, e[i]);
  red[t] = mx; __syncthreads();
  for (int o = 128; o > 0; o >>= 1) { if (t < o) red[t] = fmaxf(red[t], red[t + o]); __syncthreads(); }
  if (t == 0) stat[0] = red[0];
  __syncthreads();
  mx = stat[0];
  float sm = 0.f;
  for (int i = s0 + t; i < s1; i += 256) sm += expf(e[i] - mx);
  red[t] = sm; __syncthreads();
  for (int o = 128; o > 0; o >>= 1) { if (t < o) red[t] += red[t + o]; __syncthreads(); }
  if (t == 0) stat[1] = red[0];
  __syncthreads();
  float inv = 1.f / stat[1];
  for (int i = s0 + t; i < s1; i += 256) e[i] = expf(e[i] - mx) * inv;
}

__global__ void rpool(const float* __restrict__ x, const float* __restrict__ a,
                      const float* __restrict__ h, const int* __restrict__ off,
                      float* __restrict__ qs) {
  int b = blockIdx.x;
  int s0 = off[b], s1 = off[b + 1];
  for (int c = threadIdx.x; c < D6; c += 256) {
    float r = 0.f;
    for (int i = s0; i < s1; ++i) r += a[i] * x[(size_t)i * D6 + c];
    qs[(size_t)b * QSD + D6 + c] = r;
    qs[(size_t)b * QSD + c] = h[(size_t)b * D6 + c];
  }
}

// -------------------------------------------------------------- MLP kernels
__global__ void g_prep(const float* __restrict__ q1, const float* __restrict__ q2,
                       bf16* __restrict__ g) {
  int idx = blockIdx.x * blockDim.x + threadIdx.x;
  if (idx >= NB * G4) return;
  int b = idx / G4, k = idx % G4;
  g[idx] = (bf16)((k < QSD) ? q1[(size_t)b * QSD + k] : q2[(size_t)b * QSD + (k - QSD)]);
}

__global__ void bias_relu_bf(const float* __restrict__ z, const float* __restrict__ bias,
                             bf16* __restrict__ out, int cols) {
  int idx = blockIdx.x * blockDim.x + threadIdx.x;
  if (idx >= NB * cols) return;
  int c = idx % cols;
  out[idx] = (bf16)fmaxf(z[idx] + bias[c], 0.f);
}

__global__ void final_kernel(const float* __restrict__ z2, const float* __restrict__ bp2,
                             const float* __restrict__ Wp3, const float* __restrict__ bp3,
                             float* __restrict__ out) {
  int b = blockIdx.x * blockDim.x + threadIdx.x;
  if (b >= NB) return;
  float s = bp3[0];
  for (int j = 0; j < 128; ++j) {
    float v = fmaxf(z2[(size_t)b * 128 + j] + bp2[j], 0.f);
    s += v * Wp3[j];
  }
  out[b] = 1.f / (1.f + expf(-s));
}

// ------------------------------------------------------------------- launch
extern "C" void kernel_launch(void* const* d_in, const int* in_sizes, int n_in,
                              void* d_out, int out_size, void* d_ws, size_t ws_size,
                              hipStream_t stream) {
  (void)in_sizes; (void)n_in; (void)out_size; (void)ws_size;
  const float* feat1 = (const float*)d_in[0];
  const float* feat2 = (const float*)d_in[1];
  const float* ea1 = (const float*)d_in[2];
  const float* ea2 = (const float*)d_in[3];
  const float* Ws[3] = {(const float*)d_in[4], (const float*)d_in[6], (const float*)d_in[8]};
  const float* bs[3] = {(const float*)d_in[5], (const float*)d_in[7], (const float*)d_in[9]};
  const float* Wih = (const float*)d_in[10]; const float* bih = (const float*)d_in[11];
  const float* Whh = (const float*)d_in[12]; const float* bhh = (const float*)d_in[13];
  const float* Wp1 = (const float*)d_in[14]; const float* bp1 = (const float*)d_in[15];
  const float* Wp2 = (const float*)d_in[16]; const float* bp2 = (const float*)d_in[17];
  const float* Wp3 = (const float*)d_in[18]; const float* bp3 = (const float*)d_in[19];
  const int* ei1 = (const int*)d_in[20];
  const int* ei2 = (const int*)d_in[21];
  const int* batch1 = (const int*)d_in[22];
  const int* batch2 = (const int*)d_in[23];
  float* out = (float*)d_out;

  // workspace carve-up
  char* base = (char*)d_ws;
  size_t cur = 0;
  auto alloc = [&](size_t bytes) -> char* {
    char* p = base + cur;
    cur = (cur + bytes + 255) & ~(size_t)255;
    return p;
  };
  int*   off1  = (int*)alloc((NB + 1) * sizeof(int));
  int*   off2  = (int*)alloc((NB + 1) * sizeof(int));
  float* dinv1 = (float*)alloc((size_t)N_NODES * sizeof(float));
  float* dinv2 = (float*)alloc((size_t)N_NODES * sizeof(float));
  float* xw    = (float*)alloc((size_t)N_NODES * NHID * sizeof(float));
  float* h1    = (float*)alloc((size_t)N_NODES * H3 * sizeof(float));
  float* h2    = (float*)alloc((size_t)N_NODES * H3 * sizeof(float));
  bf16*  nsbf  = (bf16*)alloc((size_t)N_NODES * H3 * sizeof(bf16));
  bf16*  nvbf  = (bf16*)alloc((size_t)N_NODES * H3 * sizeof(bf16));
  float* f1    = (float*)alloc((size_t)N_NODES * D6 * sizeof(float));
  float* f2    = (float*)alloc((size_t)N_NODES * D6 * sizeof(float));
  bf16*  wcat  = (bf16*)alloc((size_t)G4 * KCAT * sizeof(bf16));
  bf16*  wp1b  = (bf16*)alloc((size_t)256 * G4 * sizeof(bf16));
  bf16*  wp2b  = (bf16*)alloc((size_t)128 * 256 * sizeof(bf16));
  float* hS[2];  float* cS[2];  float* qs[2];  bf16* qhb[2];  float* gts[2];
  for (int s = 0; s < 2; ++s) {
    hS[s]  = (float*)alloc((size_t)NB * D6 * sizeof(float));
    cS[s]  = (float*)alloc((size_t)NB * D6 * sizeof(float));
    qs[s]  = (float*)alloc((size_t)NB * QSD * sizeof(float));
    qhb[s] = (bf16*)alloc((size_t)NB * KCAT * sizeof(bf16));
    gts[s] = (float*)alloc((size_t)NB * G4 * sizeof(float));
  }
  float* ebuf = (float*)alloc((size_t)N_NODES * sizeof(float));
  bf16*  gbf  = (bf16*)alloc((size_t)NB * G4 * sizeof(bf16));
  float* z1   = (float*)alloc((size_t)NB * 256 * sizeof(float));
  bf16*  z1b  = (bf16*)alloc((size_t)NB * 256 * sizeof(bf16));
  float* z2   = (float*)alloc((size_t)NB * 128 * sizeof(float));

  // ----- graph-id offsets (batch arrays are sorted)
  offsets_kernel<<<1, 512, 0, stream>>>(batch1, N_NODES, off1);
  offsets_kernel<<<1, 512, 0, stream>>>(batch2, N_NODES, off2);

  // ----- degree / D^{-1/2} (self-loop weight 1)
  fill_f32<<<(N_NODES + 255) / 256, 256, 0, stream>>>(dinv1, 1.0f, N_NODES);
  fill_f32<<<(N_NODES + 255) / 256, 256, 0, stream>>>(dinv2, 1.0f, N_NODES);
  deg_acc<<<(N_EDGES + 255) / 256, 256, 0, stream>>>(ei1, ea1, dinv1, N_EDGES);
  deg_acc<<<(N_EDGES + 255) / 256, 256, 0, stream>>>(ei2, ea2, dinv2, N_EDGES);
  rsqrt_inplace<<<(N_NODES + 255) / 256, 256, 0, stream>>>(dinv1, N_NODES);
  rsqrt_inplace<<<(N_NODES + 255) / 256, 256, 0, stream>>>(dinv2, N_NODES);

  // ----- one-time weight conversions to bf16
  wcat_kernel<<<(G4 * KCAT + 255) / 256, 256, 0, stream>>>(Wih, Whh, wcat);
  f2bf_kernel<<<(256 * G4 + 255) / 256, 256, 0, stream>>>(Wp1, wp1b, 256 * G4);
  f2bf_kernel<<<(128 * 256 + 255) / 256, 256, 0, stream>>>(Wp2, wp2b, 128 * 256);

  // ----- 3 GCN layers per graph set, outputs concatenated into h{1,2}[N][384]
  for (int g = 0; g < 2; ++g) {
    const float* feat = g ? feat2 : feat1;
    const float* ea = g ? ea2 : ea1;
    const int* ei = g ? ei2 : ei1;
    float* dinv = g ? dinv2 : dinv1;
    float* h = g ? h2 : h1;
    for (int l = 0; l < 3; ++l) {
      const float* X = (l == 0) ? feat : (h + (l - 1) * NHID);
      int ldx = (l == 0) ? F_INC : H3;
      int K = (l == 0) ? F_INC : NHID;
      gemm_xw<<<N_NODES, NHID, 0, stream>>>(X, ldx, Ws[l], K, xw);
      float* hcol = h + l * NHID;
      gcn_self<<<N_NODES, NHID, 0, stream>>>(xw, dinv, hcol, H3);
      gcn_edge<<<N_EDGES, NHID, 0, stream>>>(ei, ea, xw, dinv, hcol, H3, N_EDGES);
      gcn_finish<<<N_NODES, NHID, 0, stream>>>(hcol, H3, bs[l]);
    }
  }

  // ----- L2 norm + bf16 copies; zero interaction halves of f1/f2
  l2norm_kernel<<<N_NODES, 128, 0, stream>>>(h1, f1, nsbf);
  l2norm_kernel<<<N_NODES, 128, 0, stream>>>(h2, f2, nvbf);

  // ----- masked cosine-similarity interaction: block-diagonal per graph (WMMA)
  size_t smem = (size_t)(2 * CH * H3 + CH * CH) * sizeof(bf16);   // ~104 KB
  masked_sim_kernel<<<NB, 256, smem, stream>>>(nsbf, nvbf, off1, off2, f1, f2);

  // ----- Set2Set (2 steps) per set
  for (int s = 0; s < 2; ++s) {
    const float* x = s ? f2 : f1;
    const int* batch = s ? batch2 : batch1;
    const int* off = s ? off2 : off1;
    fill_f32<<<(NB * D6 + 255) / 256, 256, 0, stream>>>(hS[s], 0.f, NB * D6);
    fill_f32<<<(NB * D6 + 255) / 256, 256, 0, stream>>>(cS[s], 0.f, NB * D6);
    fill_f32<<<(NB * QSD + 255) / 256, 256, 0, stream>>>(qs[s], 0.f, NB * QSD);
    for (int step = 0; step < 2; ++step) {
      qh_prep<<<(NB * KCAT + 255) / 256, 256, 0, stream>>>(qs[s], hS[s], qhb[s]);
      wmma_gemm_abt<<<dim3(NB / 16, G4 / 128), 256, 0, stream>>>(
          qhb[s], KCAT, wcat, KCAT, gts[s], G4, KCAT);
      lstm_update<<<(NB * D6 + 255) / 256, 256, 0, stream>>>(gts[s], bih, bhh, hS[s], cS[s]);
      e_dot<<<N_NODES, 256, 0, stream>>>(x, hS[s], batch, ebuf);
      seg_softmax<<<NB, 256, 0, stream>>>(ebuf, off);
      rpool<<<NB, 256, 0, stream>>>(x, ebuf, hS[s], off, qs[s]);
    }
  }

  // ----- final MLP: relu(g@Wp1^T+bp1) -> relu(.@Wp2^T+bp2) -> sigmoid(.@Wp3^T+bp3)
  g_prep<<<(NB * G4 + 255) / 256, 256, 0, stream>>>(qs[0], qs[1], gbf);
  wmma_gemm_abt<<<dim3(NB / 16, 256 / 128), 256, 0, stream>>>(gbf, G4, wp1b, G4, z1, 256, G4);
  bias_relu_bf<<<(NB * 256 + 255) / 256, 256, 0, stream>>>(z1, bp1, z1b, 256);
  wmma_gemm_abt<<<dim3(NB / 16, 128 / 128), 256, 0, stream>>>(z1b, 256, wp2b, 256, z2, 128, 256);
  final_kernel<<<1, 256, 0, stream>>>(z2, bp2, Wp3, bp3, out);
}
